// RowSelfAttention_85564338471246
// MI455X (gfx1250) — matrix operand order repaired and checked by reading, not compiled, not run on hardware
//
#include <hip/hip_runtime.h>
#include <hip/hip_bf16.h>

// Problem constants (R,C,B,E) = (128,256,1,768), H=12, D=64
#define R_   128
#define C_   256
#define E_   768
#define H_   12
#define D_   64
#define RC_  (R_ * C_)   // 32768 rows of x
#define RD_  (R_ * D_)   // 8192 = tied-attention contraction length per head

// q scaling = D^-0.5 / sqrt(R) = 0.125 / sqrt(128)
#define QSCALE 0.01104854346f

// LDS tile strides (in ushorts), padded to dodge bank conflicts.
// 40 ushorts = 80B rows -> every row 16B-aligned (80 = 5*16) for b128 async DMA.
#define LDSAS 40
#define LDSBS 40

typedef __attribute__((ext_vector_type(16))) __bf16 v16bf;
typedef __attribute__((ext_vector_type(8)))  float  v8f;

union Frag16 { v16bf v; unsigned int u[8]; };

static __device__ __forceinline__ unsigned short f2bf(float f) {
  union { float f; unsigned int u; } x; x.f = f;
  unsigned int r = x.u + 0x7FFFu + ((x.u >> 16) & 1u);  // round-to-nearest-even
  return (unsigned short)(r >> 16);
}

// ---------------------------------------------------------------------------
// CDNA5 async global->LDS DMA (ASYNCcnt-tracked; no VGPR round trip).
// LDS destination address = low 32 bits of the generic pointer (ISA 10.2:
// generic->LDS keeps addr[31:0]); global address in a 64-bit VGPR pair.
// Async loads complete IN ORDER, so s_wait_asynccnt N ends the oldest ops
// first -> cheap double buffering.
// ---------------------------------------------------------------------------
static __device__ __forceinline__ void async_copy_b128(const void* g, void* lds) {
  unsigned int ldsaddr = (unsigned int)(unsigned long long)lds;
  asm volatile("global_load_async_to_lds_b128 %0, %1, off"
               :: "v"(ldsaddr), "v"((unsigned long long)g)
               : "memory");
}
static __device__ __forceinline__ void wait_async0() {
  asm volatile("s_wait_asynccnt 0x0" ::: "memory");
}
static __device__ __forceinline__ void wait_async4() {   // keep newest 4 in flight
  asm volatile("s_wait_asynccnt 0x4" ::: "memory");
}

// ---------------------------------------------------------------------------
// Fragment loads from LDS for the 16-bit WMMA operands, in this code's contiguous k order (k = 16*(l>>4) + s).
// A (16x32 bf16): lane l holds row M=l&15; K-halves selected by l>>4.
// B (32x16 bf16): lane l holds col N=l&15; ldsB stored column-major [n][k]
// so a lane's 16 shorts are contiguous.
// ---------------------------------------------------------------------------
static __device__ __forceinline__ v16bf load_afrag(const unsigned short* ldsA, int rowBase, int lane) {
  Frag16 f;
  int m  = rowBase + (lane & 15);
  int kb = (lane >> 4) * 16;
  const unsigned int* p0 = (const unsigned int*)(ldsA + m * LDSAS + kb);
  const unsigned int* p1 = (const unsigned int*)(ldsA + m * LDSAS + 8 + kb);
  f.u[0] = p0[0]; f.u[1] = p0[1]; f.u[2] = p0[2]; f.u[3] = p0[3];
  f.u[4] = p1[0]; f.u[5] = p1[1]; f.u[6] = p1[2]; f.u[7] = p1[3];
  return f.v;
}

static __device__ __forceinline__ v16bf load_bfrag(const unsigned short* ldsB, int colBase, int lane) {
  Frag16 f;
  int n  = colBase + (lane & 15);
  int kb = (lane >> 4) * 16;
  const unsigned int* p = (const unsigned int*)(ldsB + n * LDSBS + kb);
#pragma unroll
  for (int i = 0; i < 8; ++i) f.u[i] = p[i];
  return f.v;
}

// One K-step (K=32) for a wave's 32x64 output region: 2 A frags x 4 B frags = 8 WMMAs
static __device__ __forceinline__ void mma_step(const unsigned short* ldsA, const unsigned short* ldsB,
                                                int waveM, int waveN, int lane, v8f acc[2][4]) {
  v16bf a0 = load_afrag(ldsA, waveM,      lane);
  v16bf a1 = load_afrag(ldsA, waveM + 16, lane);
  v16bf b[4];
#pragma unroll
  for (int j = 0; j < 4; ++j) b[j] = load_bfrag(ldsB, waveN + j * 16, lane);
#pragma unroll
  for (int j = 0; j < 4; ++j) {
    acc[0][j] = __builtin_amdgcn_wmma_f32_16x16x32_bf16(false, a0, false, b[j], (short)0, acc[0][j], false, false);
    acc[1][j] = __builtin_amdgcn_wmma_f32_16x16x32_bf16(false, a1, false, b[j], (short)0, acc[1][j], false, false);
  }
}

// ---------------------------------------------------------------------------
// Tile staging helpers (128x32 A tile / 32x128 B tile per K-step, 256 threads)
// ---------------------------------------------------------------------------

// Contiguous 128x32 bf16 tile, row-major source -> row-major LDS, async DMA (2 ops/thread).
static __device__ __forceinline__ void stageA_async(const unsigned short* A, int lda, int rowBase, int k0,
                                                    unsigned short* ldsA, int tid) {
#pragma unroll
  for (int c = 0; c < 2; ++c) {
    int ch  = c * 256 + tid;            // 512 chunks of 8 bf16 (16B)
    int row = ch >> 2;
    int kc  = (ch & 3) * 8;
    async_copy_b128(A + (size_t)(rowBase + row) * lda + k0 + kc,
                    ldsA + row * LDSAS + kc);
  }
}

// Same shape but indexed by the WMMA "n" (column) dimension: rows of Bt are
// contiguous in k, landing directly in column-major ldsB[n][k]. Async DMA.
static __device__ __forceinline__ void stageBt_async(const unsigned short* Bt, int ldt, int colBase, int k0,
                                                     unsigned short* ldsB, int tid) {
#pragma unroll
  for (int c = 0; c < 2; ++c) {
    int ch = c * 256 + tid;
    int n  = ch >> 2;
    int kc = (ch & 3) * 8;
    async_copy_b128(Bt + (size_t)(colBase + n) * ldt + k0 + kc,
                    ldsB + n * LDSBS + kc);
  }
}

// Pipelined row-major B staging: global->VGPR now, VGPR->LDS (transposed) later.
static __device__ __forceinline__ void loadB_rowmajor(const unsigned short* B, int ldb, int colBase, int k0,
                                                      int tid, uint4 d[2]) {
#pragma unroll
  for (int c = 0; c < 2; ++c) {
    int ch = c * 256 + tid;             // 512 chunks of 8 bf16
    int k  = ch >> 4;
    int nc = (ch & 15) * 8;
    d[c] = *(const uint4*)(B + (size_t)(k0 + k) * ldb + colBase + nc);
  }
}
static __device__ __forceinline__ void storeB_lds(unsigned short* ldsB, int tid, const uint4 d[2]) {
#pragma unroll
  for (int c = 0; c < 2; ++c) {
    int ch = c * 256 + tid;
    int k  = ch >> 4;
    int nc = (ch & 15) * 8;
    unsigned int w[4] = { d[c].x, d[c].y, d[c].z, d[c].w };
#pragma unroll
    for (int e = 0; e < 4; ++e) {
      ldsB[(nc + 2 * e + 0) * LDSBS + k] = (unsigned short)(w[e] & 0xFFFFu);
      ldsB[(nc + 2 * e + 1) * LDSBS + k] = (unsigned short)(w[e] >> 16);
    }
  }
}

// Unpipelined variants (context kernel)
static __device__ __forceinline__ void stageB_rowmajor(const unsigned short* B, int ldb, int colBase, int k0,
                                                       unsigned short* ldsB, int tid) {
  uint4 d[2];
  loadB_rowmajor(B, ldb, colBase, k0, tid, d);
  storeB_lds(ldsB, tid, d);
}

// A source f32 (attention probs) -> bf16 LDS tile (conversion; VGPR path).
static __device__ __forceinline__ void stageA_f32(const float* A, int lda, int rowBase, int k0,
                                                  unsigned short* ldsA, int tid) {
#pragma unroll
  for (int e = 0; e < 16; ++e) {
    int idx = e * 256 + tid;            // 4096 elements
    int row = idx >> 5;
    int kc  = idx & 31;
    ldsA[row * LDSAS + kc] = f2bf(A[(size_t)(rowBase + row) * lda + k0 + kc]);
  }
}

// ---------------------------------------------------------------------------
// Kernels
// ---------------------------------------------------------------------------
__global__ void f2bf_kernel(const float* __restrict__ src, unsigned short* __restrict__ dst, int n) {
  int i = blockIdx.x * blockDim.x + threadIdx.x;
  if (i < n) dst[i] = f2bf(src[i]);
}

// Y = Xb @ W + b (scaled for Q); scatter bf16 into head-major [h][c][r*D+d].
// Software pipelined: async A(t+1) + global B(t+1) loads overlap WMMA(t);
// single barrier per K-step.
__global__ __launch_bounds__(256) void qkv_gemm_kernel(
    const unsigned short* __restrict__ Xb,
    const unsigned short* __restrict__ Wqb, const unsigned short* __restrict__ Wkb,
    const unsigned short* __restrict__ Wvb,
    const float* __restrict__ bq, const float* __restrict__ bk, const float* __restrict__ bv,
    unsigned short* __restrict__ Qh, unsigned short* __restrict__ Kh, unsigned short* __restrict__ Vh) {
  __shared__ __align__(16) unsigned short ldsA[2][128 * LDSAS];
  __shared__ __align__(16) unsigned short ldsB[2][128 * LDSBS];
  int tid = threadIdx.x, lane = tid & 31, wave = tid >> 5;
  int waveM = (wave >> 1) * 32, waveN = (wave & 1) * 64;
  int tileN = blockIdx.x * 128;
  int tileM = blockIdx.y * 128;
  int which = blockIdx.z;
  const unsigned short* W    = (which == 0) ? Wqb : (which == 1) ? Wkb : Wvb;
  const float*          bias = (which == 0) ? bq  : (which == 1) ? bk  : bv;
  unsigned short*       dst  = (which == 0) ? Qh  : (which == 1) ? Kh  : Vh;
  float scale = (which == 0) ? QSCALE : 1.0f;

  v8f acc[2][4];
  for (int i = 0; i < 2; ++i) for (int j = 0; j < 4; ++j) for (int e = 0; e < 8; ++e) acc[i][j][e] = 0.f;

  uint4 breg[2];
  // prologue: tile 0
  stageA_async(Xb, E_, tileM, 0, ldsA[0], tid);
  loadB_rowmajor(W, E_, tileN, 0, tid, breg);
  storeB_lds(ldsB[0], tid, breg);
  wait_async0();
  __syncthreads();

  const int NT = E_ / 32;   // 24
  for (int kt = 0; kt < NT; ++kt) {
    int cur = kt & 1, nxt = cur ^ 1;
    bool more = (kt + 1 < NT);
    if (more) {
      stageA_async(Xb, E_, tileM, (kt + 1) * 32, ldsA[nxt], tid);   // DMA in flight during WMMA
      loadB_rowmajor(W, E_, tileN, (kt + 1) * 32, tid, breg);       // global loads in flight
    }
    mma_step(ldsA[cur], ldsB[cur], waveM, waveN, lane, acc);
    if (more) {
      storeB_lds(ldsB[nxt], tid, breg);
      wait_async0();
    }
    __syncthreads();
  }

#pragma unroll
  for (int i = 0; i < 2; ++i)
#pragma unroll
    for (int j = 0; j < 4; ++j)
#pragma unroll
      for (int vg = 0; vg < 8; ++vg) {
        int row = tileM + waveM + i * 16 + vg + ((lane >> 4) * 8);   // rc index
        int col = tileN + waveN + j * 16 + (lane & 15);              // e index
        float v = (acc[i][j][vg] + bias[col]) * scale;
        int r = row >> 8, cc = row & 255;   // row = r*C + c
        int h = col >> 6, d  = col & 63;    // col = h*D + d
        dst[(size_t)(h * C_ + cc) * RD_ + r * D_ + d] = f2bf(v);
      }
}

// S_h = Q_h @ K_h^T, K-dim = R*D = 8192; double-buffered pure-async DMA.
// Issue tile t+1 (4 DMAs/thread), then s_wait_asynccnt 4: the in-order
// ASYNCcnt retires tile t's DMAs while t+1's stay in flight behind the WMMAs.
__global__ __launch_bounds__(256) void scores_kernel(
    const unsigned short* __restrict__ Qh, const unsigned short* __restrict__ Kh,
    float* __restrict__ probs) {
  __shared__ __align__(16) unsigned short ldsA[2][128 * LDSAS];
  __shared__ __align__(16) unsigned short ldsB[2][128 * LDSBS];
  int tid = threadIdx.x, lane = tid & 31, wave = tid >> 5;
  int waveM = (wave >> 1) * 32, waveN = (wave & 1) * 64;
  int tileN = blockIdx.x * 128;   // j
  int tileM = blockIdx.y * 128;   // i
  int h = blockIdx.z;
  const unsigned short* Q = Qh + (size_t)h * C_ * RD_;
  const unsigned short* K = Kh + (size_t)h * C_ * RD_;

  v8f acc[2][4];
  for (int i = 0; i < 2; ++i) for (int j = 0; j < 4; ++j) for (int e = 0; e < 8; ++e) acc[i][j][e] = 0.f;

  // prologue: tile 0
  stageA_async(Q, RD_, tileM, 0, ldsA[0], tid);
  stageBt_async(K, RD_, tileN, 0, ldsB[0], tid);

  const int NT = RD_ / 32;   // 256
  for (int kt = 0; kt < NT; ++kt) {
    int cur = kt & 1, nxt = cur ^ 1;
    if (kt + 1 < NT) {
      stageA_async(Q, RD_, tileM, (kt + 1) * 32, ldsA[nxt], tid);
      stageBt_async(K, RD_, tileN, (kt + 1) * 32, ldsB[nxt], tid);
      wait_async4();   // tile kt landed; tile kt+1 still in flight
    } else {
      wait_async0();
    }
    __syncthreads();
    mma_step(ldsA[cur], ldsB[cur], waveM, waveN, lane, acc);
    __syncthreads();
  }

#pragma unroll
  for (int i = 0; i < 2; ++i)
#pragma unroll
    for (int j = 0; j < 4; ++j)
#pragma unroll
      for (int vg = 0; vg < 8; ++vg) {
        int row = tileM + waveM + i * 16 + vg + ((lane >> 4) * 8);
        int col = tileN + waveN + j * 16 + (lane & 15);
        probs[((size_t)h * C_ + row) * C_ + col] = acc[i][j][vg];
      }
}

// Row softmax over j, in place; one block per (h,i) row of 256
__global__ __launch_bounds__(256) void softmax_kernel(float* __restrict__ probs) {
  __shared__ float red[256];
  int row = blockIdx.x, tid = threadIdx.x;
  float v = probs[(size_t)row * 256 + tid];
  red[tid] = v; __syncthreads();
  for (int s = 128; s > 0; s >>= 1) { if (tid < s) red[tid] = fmaxf(red[tid], red[tid + s]); __syncthreads(); }
  float m = red[0]; __syncthreads();
  float e = __expf(v - m);
  red[tid] = e; __syncthreads();
  for (int s = 128; s > 0; s >>= 1) { if (tid < s) red[tid] += red[tid + s]; __syncthreads(); }
  probs[(size_t)row * 256 + tid] = e / red[0];
}

// Ctx_h = P_h(256x256) @ V_h(256x8192); scatter bf16 into [r*C+i][h*D+d]
__global__ __launch_bounds__(256) void context_kernel(
    const float* __restrict__ probs, const unsigned short* __restrict__ Vh,
    unsigned short* __restrict__ Ctx) {
  __shared__ __align__(16) unsigned short ldsA[128 * LDSAS];
  __shared__ __align__(16) unsigned short ldsB[128 * LDSBS];
  int tid = threadIdx.x, lane = tid & 31, wave = tid >> 5;
  int waveM = (wave >> 1) * 32, waveN = (wave & 1) * 64;
  int tileN = blockIdx.x * 128;   // t = r*D + d
  int tileM = blockIdx.y * 128;   // i
  int h = blockIdx.z;
  const float*          P = probs + (size_t)h * C_ * C_;
  const unsigned short* V = Vh + (size_t)h * C_ * RD_;

  v8f acc[2][4];
  for (int i = 0; i < 2; ++i) for (int j = 0; j < 4; ++j) for (int e = 0; e < 8; ++e) acc[i][j][e] = 0.f;

  for (int k0 = 0; k0 < C_; k0 += 32) {
    stageA_f32(P, C_, tileM, k0, ldsA, tid);
    if (k0 + 32 < C_)
      __builtin_prefetch(V + (size_t)(k0 + 32 + (tid >> 4)) * RD_ + tileN + (tid & 15) * 8, 0, 1);
    stageB_rowmajor(V, RD_, tileN, k0, ldsB, tid);
    __syncthreads();
    mma_step(ldsA, ldsB, waveM, waveN, lane, acc);
    __syncthreads();
  }

#pragma unroll
  for (int i = 0; i < 2; ++i)
#pragma unroll
    for (int j = 0; j < 4; ++j)
#pragma unroll
      for (int vg = 0; vg < 8; ++vg) {
        int row = tileM + waveM + i * 16 + vg + ((lane >> 4) * 8);   // i
        int col = tileN + waveN + j * 16 + (lane & 15);              // t
        int r = col >> 6, d = col & 63;
        Ctx[(size_t)(r * C_ + row) * E_ + h * D_ + d] = f2bf(acc[i][j][vg]);
      }
}

// Out = Ctx @ Wo + bo, f32 output row-major; same pipeline as qkv.
__global__ __launch_bounds__(256) void outproj_kernel(
    const unsigned short* __restrict__ Ctx, const unsigned short* __restrict__ Wob,
    const float* __restrict__ bo, float* __restrict__ out) {
  __shared__ __align__(16) unsigned short ldsA[2][128 * LDSAS];
  __shared__ __align__(16) unsigned short ldsB[2][128 * LDSBS];
  int tid = threadIdx.x, lane = tid & 31, wave = tid >> 5;
  int waveM = (wave >> 1) * 32, waveN = (wave & 1) * 64;
  int tileN = blockIdx.x * 128;
  int tileM = blockIdx.y * 128;

  v8f acc[2][4];
  for (int i = 0; i < 2; ++i) for (int j = 0; j < 4; ++j) for (int e = 0; e < 8; ++e) acc[i][j][e] = 0.f;

  uint4 breg[2];
  stageA_async(Ctx, E_, tileM, 0, ldsA[0], tid);
  loadB_rowmajor(Wob, E_, tileN, 0, tid, breg);
  storeB_lds(ldsB[0], tid, breg);
  wait_async0();
  __syncthreads();

  const int NT = E_ / 32;   // 24
  for (int kt = 0; kt < NT; ++kt) {
    int cur = kt & 1, nxt = cur ^ 1;
    bool more = (kt + 1 < NT);
    if (more) {
      stageA_async(Ctx, E_, tileM, (kt + 1) * 32, ldsA[nxt], tid);
      loadB_rowmajor(Wob, E_, tileN, (kt + 1) * 32, tid, breg);
    }
    mma_step(ldsA[cur], ldsB[cur], waveM, waveN, lane, acc);
    if (more) {
      storeB_lds(ldsB[nxt], tid, breg);
      wait_async0();
    }
    __syncthreads();
  }

#pragma unroll
  for (int i = 0; i < 2; ++i)
#pragma unroll
    for (int j = 0; j < 4; ++j)
#pragma unroll
      for (int vg = 0; vg < 8; ++vg) {
        int row = tileM + waveM + i * 16 + vg + ((lane >> 4) * 8);
        int col = tileN + waveN + j * 16 + (lane & 15);
        out[(size_t)row * E_ + col] = acc[i][j][vg] + bo[col];
      }
}

// ---------------------------------------------------------------------------
extern "C" void kernel_launch(void* const* d_in, const int* in_sizes, int n_in,
                              void* d_out, int out_size, void* d_ws, size_t ws_size,
                              hipStream_t stream) {
  (void)in_sizes; (void)n_in; (void)out_size; (void)ws_size;
  const float* x  = (const float*)d_in[0];
  const float* wq = (const float*)d_in[1];
  const float* bq = (const float*)d_in[2];
  const float* wk = (const float*)d_in[3];
  const float* bk = (const float*)d_in[4];
  const float* wv = (const float*)d_in[5];
  const float* bv = (const float*)d_in[6];
  const float* wo = (const float*)d_in[7];
  const float* bo = (const float*)d_in[8];

  float* out   = (float*)d_out;
  float* probs = out + (size_t)RC_ * E_;   // attn_probs region (H*C*C f32)

  // workspace layout (bf16 = ushort)
  unsigned short* Xb  = (unsigned short*)d_ws;          // RC x E
  unsigned short* Wqb = Xb  + (size_t)RC_ * E_;
  unsigned short* Wkb = Wqb + (size_t)E_ * E_;
  unsigned short* Wvb = Wkb + (size_t)E_ * E_;
  unsigned short* Wob = Wvb + (size_t)E_ * E_;
  unsigned short* Qh  = Wob + (size_t)E_ * E_;          // [h][c][r*D+d]
  unsigned short* Kh  = Qh  + (size_t)H_ * C_ * RD_;
  unsigned short* Vh  = Kh  + (size_t)H_ * C_ * RD_;
  unsigned short* Ctx = Vh  + (size_t)H_ * C_ * RD_;    // [r*C+c][e]

  f2bf_kernel<<<(RC_ * E_ + 255) / 256, 256, 0, stream>>>(x,  Xb,  RC_ * E_);
  f2bf_kernel<<<(E_ * E_ + 255) / 256, 256, 0, stream>>>(wq, Wqb, E_ * E_);
  f2bf_kernel<<<(E_ * E_ + 255) / 256, 256, 0, stream>>>(wk, Wkb, E_ * E_);
  f2bf_kernel<<<(E_ * E_ + 255) / 256, 256, 0, stream>>>(wv, Wvb, E_ * E_);
  f2bf_kernel<<<(E_ * E_ + 255) / 256, 256, 0, stream>>>(wo, Wob, E_ * E_);

  qkv_gemm_kernel<<<dim3(E_ / 128, RC_ / 128, 3), 256, 0, stream>>>(
      Xb, Wqb, Wkb, Wvb, bq, bk, bv, Qh, Kh, Vh);

  scores_kernel<<<dim3(C_ / 128, C_ / 128, H_), 256, 0, stream>>>(Qh, Kh, probs);

  softmax_kernel<<<H_ * C_, 256, 0, stream>>>(probs);

  context_kernel<<<dim3(RD_ / 128, C_ / 128, H_), 256, 0, stream>>>(probs, Vh, Ctx);

  outproj_kernel<<<dim3(E_ / 128, RC_ / 128, 1), 256, 0, stream>>>(Ctx, Wob, bo, out);
}
